// A161_FusionResidualStabilizer_79534204387425
// MI455X (gfx1250) — matrix operand order repaired and checked by reading, not compiled, not run on hardware
//
#include <hip/hip_runtime.h>
#include <hip/hip_bf16.h>
#include <cstdint>

// Problem constants (match reference)
#define DIM      2048
#define M_TOTAL  16384        // 4 * 4096
#define M_TILE   32           // rows per workgroup (whole rows owned -> fused row norms)
#define NC       64           // N chunk per iteration (4 waves wide x 16)
#define KC       32           // K chunk staged in LDS per step (1 WMMA K-step of 32)
#define ALPHA    0.1f
#define EPS_     1e-12f

#define A_STRIDE (DIM + 8)    // bf16 elements; pad 8 (16B) -> bank rotation, 16B aligned rows
#define TILE_BYTES (KC * NC * 2)          // 4096 B per staged B tile
#define NTILES     ((DIM / KC) * (DIM / NC))  // 2048 linear tiles per matrix

typedef __attribute__((ext_vector_type(16))) __bf16 v16bf;
typedef __attribute__((ext_vector_type(8)))  __bf16 v8bf;
typedef __attribute__((ext_vector_type(8)))  float  v8f;
typedef __attribute__((ext_vector_type(4)))  uint32_t u32x4;
typedef __attribute__((ext_vector_type(8)))  int      i32x8;
typedef __attribute__((ext_vector_type(4)))  int      i32x4;

// Round-to-nearest-even f32 -> bf16, packed pair (lo in bits [15:0])
__device__ __forceinline__ uint32_t pack_bf16x2(float lo, float hi) {
  uint32_t ul = __float_as_uint(lo);
  ul += 0x7fffu + ((ul >> 16) & 1u);
  uint32_t uh = __float_as_uint(hi);
  uh += 0x7fffu + ((uh >> 16) & 1u);
  return (ul >> 16) | (uh & 0xffff0000u);
}

// ---------------------------------------------------------------------------
// TDM: 1-D contiguous tile copy global -> LDS via Tensor Data Mover.
// D# group0: count=1 | lds_addr | global_addr[31:0] | global_addr[56:32]+type=2
// D# group1: data_size=8B; tensor_dim0 = tile_dim0 = TILE_BYTES/8; dims1 = 1.
// Groups 2/3 zero (<=2D tensor). Tracked with TENSORcnt.
// 6-arg builtin form (clang-23 / therock-10.0 toolchain).
// ---------------------------------------------------------------------------
__device__ __forceinline__ void tdm_load_tile(const void* gptr, void* lptr) {
  const uint64_t ga = (uint64_t)(uintptr_t)gptr;
  const uint32_t la = (uint32_t)(uintptr_t)lptr;  // low 32 bits = LDS offset
  const uint32_t elems = TILE_BYTES / 8;          // 512 x 8-byte elements

  u32x4 g0 = { 1u,                                            // count=1
               la,                                            // lds_addr
               (uint32_t)(ga & 0xffffffffu),                  // global_addr lo
               (uint32_t)((ga >> 32) & 0x01ffffffu) | (2u << 30) }; // hi | type=2

  i32x8 g1 = { (int)(3u << 16),          // data_size = 3 (8 bytes); wg_mask=0
               (int)((elems & 0xffffu) << 16),   // tensor_dim0[15:0] @ bits 63:48
               (int)(1u << 16),          // tensor_dim0[31:16]=0 | tensor_dim1=1
               (int)((elems & 0xffffu) << 16),   // tile_dim0 @ bits 127:112
               1,                        // tile_dim1 = 1
               (int)elems,               // tensor_dim0_stride[31:0]
               0, 0 };
  i32x4 z4 = { 0, 0, 0, 0 };
  i32x8 z8 = { 0, 0, 0, 0, 0, 0, 0, 0 };
  __builtin_amdgcn_tensor_load_to_lds(g0, g1, z4, z4, z8, 0);
}

// ---------------------------------------------------------------------------
// Kernel 1: convert R (f32 row-major KxN) into bf16, pre-swizzled into the
// CDNA5 WMMA B-fragment layout. Stream layout: per 32x64 (KxN) tile of 4 KB,
// ordered nc-major then K-block-minor, so the main kernel consumes a single
// contiguous byte stream (tile T at byte T*4096). Within a tile: 4 sub-blocks
// (nsub over N16); lane L stores 16 bf16:
//   B[k0 + 16*(L/16) + t][n0 + 16*nsub + (L%16)],  t = 0..15
// i.e. the contiguous-K v_wmma_*_bf16 B order this code pairs with A (lanes 0-15: K=0..15,
// lanes 16-31: K=16..31, 2 elements per VGPR in ascending K).
// ---------------------------------------------------------------------------
__global__ void k_convert_swizzle_B(const float* __restrict__ R1,
                                    const float* __restrict__ R2,
                                    uint32_t* __restrict__ O1,
                                    uint32_t* __restrict__ O2) {
  const int mat = blockIdx.x >> 10;           // 1024 blocks per matrix
  const int b   = blockIdx.x & 1023;          // 8 KB super-tile = 2 stream tiles
  const int nc  = b >> 5;
  const int ks  = b & 31;                     // pair of K-blocks (KC=32 each)
  const int tid = threadIdx.x;
  const int sb  = tid >> 5;                   // sub-block 0..7
  const int L   = tid & 31;
  const int j    = sb >> 2;                   // which KC=32 block in the pair
  const int nsub = sb & 3;

  const float*  R   = (mat == 0) ? R1 : R2;
  uint32_t*     dst = (mat == 0) ? O1 : O2;

  const int krow = ks * 64 + 32 * j + 16 * (L >> 4);
  const int col  = nc * NC + 16 * nsub + (L & 15);
  const float* src = R + (size_t)krow * DIM + col;

  uint32_t v[8];
#pragma unroll
  for (int i = 0; i < 8; ++i) {
    float a  = src[(size_t)(2 * i)     * DIM];
    float bb = src[(size_t)(2 * i + 1) * DIM];
    v[i] = pack_bf16x2(a, bb);
  }
  // stream tile index = nc*64 + (2*ks + j); 32-byte chunk (nsub*32+L) inside
  const size_t tileT = (size_t)nc * 64 + 2 * ks + j;
  uint4* o = reinterpret_cast<uint4*>(dst) + (tileT * 256 + (nsub * 32 + L) * 2);
  o[0] = make_uint4(v[0], v[1], v[2], v[3]);
  o[1] = make_uint4(v[4], v[5], v[6], v[7]);
}

// ---------------------------------------------------------------------------
// Kernel 2: fused  y = normalize( x + a*( s*(x@R1) + tanh(s*(x@R2)) ) )
// with s = 1 - 1/(||x||+eps) per row.  One WG owns 32 full rows; B tiles are
// double-buffered in LDS and filled by the Tensor Data Mover (wave 0 issues,
// TENSORcnt + workgroup barrier gate the consumers).
// ---------------------------------------------------------------------------
__global__ void __launch_bounds__(256)
k_fused_stabilizer(const float* __restrict__ x,
                   const uint32_t* __restrict__ B1swz,
                   const uint32_t* __restrict__ B2swz,
                   float* __restrict__ out) {
  __shared__ __align__(16) uint16_t Abuf[M_TILE][A_STRIDE];   // x tile, bf16 bits
  __shared__ __align__(32) uint8_t  Bb[2][2][TILE_BYTES];     // [buf][mat], 16 KB
  __shared__ float rowsum[M_TILE];
  __shared__ float s_scale[M_TILE];
  __shared__ float ynorm2[M_TILE];
  __shared__ float invn[M_TILE];

  const int tid  = threadIdx.x;
  const int row0 = blockIdx.x * M_TILE;
  const uint8_t* w1 = reinterpret_cast<const uint8_t*>(B1swz);
  const uint8_t* w2 = reinterpret_cast<const uint8_t*>(B2swz);

  // Kick off the DMA for the first tile pair immediately; it overlaps with
  // the whole phase-1 row-norm pass.
  if (tid < 32) {
    tdm_load_tile(w1, &Bb[0][0][0]);
    tdm_load_tile(w2, &Bb[0][1][0]);
  }

  // ---------------- Phase 1: load x tile, row norms, bf16 into LDS ----------
  {
    const int q = tid & 7;        // 8 threads per row
    const int r = tid >> 3;       // 0..31
    const float2* xr = reinterpret_cast<const float2*>(x + (size_t)(row0 + r) * DIM);
    float ss = 0.f;
#pragma unroll 4
    for (int it = 0; it < DIM / 16; ++it) {       // 128 iters, 2 elems each
      const int cp = it * 8 + q;                  // float2 index; c = 2*cp
      float2 v = xr[cp];
      ss += v.x * v.x + v.y * v.y;
      *reinterpret_cast<uint32_t*>(&Abuf[r][cp * 2]) = pack_bf16x2(v.x, v.y);
    }
    ss += __shfl_xor(ss, 1, 8);
    ss += __shfl_xor(ss, 2, 8);
    ss += __shfl_xor(ss, 4, 8);
    if (q == 0) rowsum[r] = ss;
    if (tid < M_TILE) ynorm2[tid] = 0.f;
  }
  __syncthreads();
  if (tid < M_TILE) {
    s_scale[tid] = 1.f - 1.f / (sqrtf(rowsum[tid]) + EPS_);
  }
  __syncthreads();

  // ---------------- Phase 2: dual GEMM over N chunks -------------------------
  const int wave  = tid >> 5;
  const int lane  = tid & 31;
  const int wm    = wave >> 2;        // 0..1  (M 16-subtile)
  const int wn    = wave & 3;         // 0..3  (N 16-subtile)
  const int half  = lane >> 4;
  const int lanen = lane & 15;

  int Tlin = 0;                       // linear tile index, contiguous stream
  for (int nc = 0; nc < DIM / NC; ++nc) {
    v8f acc1 = {};
    v8f acc2 = {};

    for (int ks = 0; ks < DIM / KC; ++ks, ++Tlin) {
      const int p = Tlin & 1;
      // wave 0: issue DMA for the NEXT tile pair into the other buffer, then
      // wait (in-order completion) until the CURRENT pair is resident.
      if (tid < 32) {
        if (Tlin + 1 < NTILES) {
          tdm_load_tile(w1 + (size_t)(Tlin + 1) * TILE_BYTES, &Bb[1 - p][0][0]);
          tdm_load_tile(w2 + (size_t)(Tlin + 1) * TILE_BYTES, &Bb[1 - p][1][0]);
          __builtin_amdgcn_s_wait_tensorcnt((short)2);
        } else {
          __builtin_amdgcn_s_wait_tensorcnt((short)0);
        }
      }
      __syncthreads();   // current buffer ready for all waves

      {
        const int k0 = ks * KC;
        // A fragment (matches B stream): lanes0-15 K=0-15, lanes16-31 K=16-31
        const uint16_t* arow = &Abuf[wm * 16 + lanen][0];
        v8bf lo = *reinterpret_cast<const v8bf*>(arow + k0 + half * 16);
        v8bf hi = *reinterpret_cast<const v8bf*>(arow + k0 + half * 16 + 8);
        v16bf afrag = __builtin_shufflevector(lo, hi,
            0, 1, 2, 3, 4, 5, 6, 7, 8, 9, 10, 11, 12, 13, 14, 15);
        // B fragments: one contiguous 32-byte chunk per lane (pre-swizzled)
        const int boff = ((wn * 32 + lane) << 5);
        v16bf bf1 = *reinterpret_cast<const v16bf*>(&Bb[p][0][0] + boff);
        v16bf bf2 = *reinterpret_cast<const v16bf*>(&Bb[p][1][0] + boff);

        acc1 = __builtin_amdgcn_wmma_f32_16x16x32_bf16(
            false, afrag, false, bf1, (short)0, acc1, false, false);
        acc2 = __builtin_amdgcn_wmma_f32_16x16x32_bf16(
            false, afrag, false, bf2, (short)0, acc2, false, false);
      }
      __syncthreads();   // readers done: buffer p may be overwritten next iter
    }

    // Epilogue for this N chunk: y = x + a*(s*p1 + tanh(s*p2)); accumulate y^2
    const int gcol = nc * NC + wn * 16 + lanen;
#pragma unroll
    for (int e = 0; e < 8; ++e) {
      const int m  = wm * 16 + half * 8 + e;      // C/D layout: VGPR e -> M
      const float sr = s_scale[m];
      const float xv = x[(size_t)(row0 + m) * DIM + gcol];
      const float yv = xv + ALPHA * (sr * acc1[e] + tanhf(sr * acc2[e]));
      out[(size_t)(row0 + m) * DIM + gcol] = yv;
      float t = yv * yv;
      t += __shfl_xor(t, 1, 16);
      t += __shfl_xor(t, 2, 16);
      t += __shfl_xor(t, 4, 16);
      t += __shfl_xor(t, 8, 16);
      if (lanen == 0) atomicAdd(&ynorm2[m], t);   // ds_add_f32
    }
  }

  // ---------------- Phase 3: in-place row normalization ----------------------
  __threadfence();          // make y stores visible to whole WG re-read
  __syncthreads();
  if (tid < M_TILE) invn[tid] = 1.f / (sqrtf(ynorm2[tid]) + EPS_);
  __syncthreads();

  float* ob = out + (size_t)row0 * DIM;
#pragma unroll 4
  for (int it = 0; it < (M_TILE * DIM) / 256; ++it) {   // 256 coalesced iters
    const int f = it * 256 + tid;
    const int r = f >> 11;                              // f / DIM
    ob[f] *= invn[r];
  }
}

// ---------------------------------------------------------------------------
extern "C" void kernel_launch(void* const* d_in, const int* in_sizes, int n_in,
                              void* d_out, int out_size, void* d_ws, size_t ws_size,
                              hipStream_t stream) {
  (void)in_sizes; (void)n_in; (void)out_size; (void)ws_size;
  const float* x  = (const float*)d_in[0];
  const float* R1 = (const float*)d_in[1];
  const float* R2 = (const float*)d_in[2];
  float* out = (float*)d_out;

  // workspace: two pre-swizzled bf16 copies of R1/R2 (8 MB each)
  uint8_t*  ws  = (uint8_t*)d_ws;
  uint32_t* O1  = (uint32_t*)(ws);
  uint32_t* O2  = (uint32_t*)(ws + (size_t)DIM * DIM * 2);

  k_convert_swizzle_B<<<2048, 256, 0, stream>>>(R1, R2, O1, O2);
  k_fused_stabilizer<<<M_TOTAL / M_TILE, 256, 0, stream>>>(x, O1, O2, out);
}
